// WrapMamba_47047071760804
// MI455X (gfx1250) — matrix operand order repaired and checked by reading, not compiled, not run on hardware
//
#include <hip/hip_runtime.h>
#include <math.h>

#define BB    4
#define LL    512
#define MM    (BB * LL)          // 2048 token rows
#define DMODEL 384
#define DIN    768
#define NST    16
#define DTR    24
#define DTR_P  64                // K padded to 64-multiple for pipelined GEMM
#define DCONV  4
#define NVOCAB 1544
#define NVOCAB_P 1600            // K padded to 64-multiple
#define NFEAT  1536
#define NLAYER 4
#define XDBL   56                // dt(24) + B(16) + C(16)
#define XDBL_P 64                // padded to N-tile multiple

typedef __attribute__((ext_vector_type(16))) __bf16        v16bf;
typedef __attribute__((ext_vector_type(8)))  float         v8f;
typedef __attribute__((ext_vector_type(4)))  unsigned int  u32x4;

union BFrag { v16bf v; u32x4 q[2]; };
union AccU  { v8f v; float f[8]; };

__device__ __forceinline__ unsigned short f2bf(float f) {
    unsigned int u = __float_as_uint(f);
    unsigned int r = (u + 0x7FFFu + ((u >> 16) & 1u)) >> 16;  // RNE
    return (unsigned short)r;
}

// ---------------------------------------------------------------------------
// fp32 -> bf16 convert with column zero-padding and source row stride.
// Ksrc == 0 acts as a pure zero-fill of rows*Kdst bf16 elements.
// ---------------------------------------------------------------------------
__global__ void k_cvt_pad(const float* __restrict__ src, unsigned short* __restrict__ dst,
                          int rows, int Ksrc, int Kdst, int srcStride) {
    int idx = blockIdx.x * blockDim.x + threadIdx.x;
    int total = rows * Kdst;
    if (idx >= total) return;
    int r = idx / Kdst, c = idx - r * Kdst;
    float v = (c < Ksrc) ? src[(size_t)r * srcStride + c] : 0.0f;
    dst[idx] = f2bf(v);
}

// ---------------------------------------------------------------------------
// RMSNorm over last dim (384), bf16 output. One block (128 thr) per row.
// ---------------------------------------------------------------------------
__global__ void k_rmsnorm_bf(const float* __restrict__ x, const float* __restrict__ w,
                             unsigned short* __restrict__ out) {
    __shared__ float red[128];
    int row = blockIdx.x, tid = threadIdx.x;
    const float* xr = x + (size_t)row * DMODEL;
    float v0 = xr[tid], v1 = xr[tid + 128], v2 = xr[tid + 256];
    red[tid] = v0 * v0 + v1 * v1 + v2 * v2;
    __syncthreads();
    for (int off = 64; off > 0; off >>= 1) {
        if (tid < off) red[tid] += red[tid + off];
        __syncthreads();
    }
    float scale = rsqrtf(red[0] * (1.0f / DMODEL) + 1e-5f);
    unsigned short* orow = out + (size_t)row * DMODEL;
    orow[tid]       = f2bf(v0 * scale * w[tid]);
    orow[tid + 128] = f2bf(v1 * scale * w[tid + 128]);
    orow[tid + 256] = f2bf(v2 * scale * w[tid + 256]);
}

// ---------------------------------------------------------------------------
// WMMA bf16 GEMM:  C[M,N] (+)= A[M,K] * W[N,K]^T + bias
// A,W bf16 row-major, K % 64 == 0, N % (TN*32) == 0, M % (TM*64) == 0.
// Block = 8 waves (4 in M x 2 in N); wave tile = (TM*16) x (TN*16).
// Double-buffered fragments: loads for k+32 overlap WMMAs on k.
// ---------------------------------------------------------------------------
template <int TM, int TN, bool ADDC>
__global__ __launch_bounds__(256) void k_gemm(
    const unsigned short* __restrict__ A, const unsigned short* __restrict__ W,
    float* __restrict__ C, const float* __restrict__ bias, int N, int K) {
    int lane = threadIdx.x & 31;
    int wave = threadIdx.x >> 5;
    int rowBase = blockIdx.y * (TM * 64) + (wave & 3) * (TM * 16);
    int colBase = blockIdx.x * (TN * 32) + (wave >> 2) * (TN * 16);
    int half = lane >> 4, l16 = lane & 15;

    const unsigned short* ar[TM];
    #pragma unroll
    for (int m = 0; m < TM; ++m)
        ar[m] = A + (size_t)(rowBase + m * 16 + l16) * K + half * 8;
    const unsigned short* wt[TN];
    #pragma unroll
    for (int t = 0; t < TN; ++t)
        wt[t] = W + (size_t)(colBase + t * 16 + l16) * K + half * 8;

    v8f acc[TM][TN];
    #pragma unroll
    for (int m = 0; m < TM; ++m)
        #pragma unroll
        for (int t = 0; t < TN; ++t) acc[m][t] = (v8f){};

    BFrag a0[TM], b0[TN], a1[TM], b1[TN];
    auto loadA = [&](BFrag (&a)[TM], int k0) {
        #pragma unroll
        for (int m = 0; m < TM; ++m) {
            a[m].q[0] = *(const u32x4*)(ar[m] + k0);       // K 0-7 / 8-15
            a[m].q[1] = *(const u32x4*)(ar[m] + k0 + 16);  // K 16-23 / 24-31
        }
    };
    auto loadB = [&](BFrag (&b)[TN], int k0) {
        #pragma unroll
        for (int t = 0; t < TN; ++t) {
            b[t].q[0] = *(const u32x4*)(wt[t] + k0);
            b[t].q[1] = *(const u32x4*)(wt[t] + k0 + 16);
        }
    };
    auto mma = [&](BFrag (&a)[TM], BFrag (&b)[TN]) {
        #pragma unroll
        for (int m = 0; m < TM; ++m)
            #pragma unroll
            for (int t = 0; t < TN; ++t)
                acc[m][t] = __builtin_amdgcn_wmma_f32_16x16x32_bf16(
                    false, a[m].v, false, b[t].v, (short)0, acc[m][t], false, false);
    };

    loadA(a0, 0); loadB(b0, 0);
    for (int k0 = 0; k0 < K; k0 += 64) {
        if (k0 + 32 < K) { loadA(a1, k0 + 32); loadB(b1, k0 + 32); }
        mma(a0, b0);
        if (k0 + 64 < K) { loadA(a0, k0 + 64); loadB(b0, k0 + 64); }
        if (k0 + 32 < K) mma(a1, b1);
    }

    // C/D layout: element e -> row base + half*8 + e, col n.
    #pragma unroll
    for (int m = 0; m < TM; ++m) {
        #pragma unroll
        for (int t = 0; t < TN; ++t) {
            AccU ac; ac.v = acc[m][t];
            int n = colBase + t * 16 + l16;
            float bv = bias ? bias[n] : 0.0f;
            #pragma unroll
            for (int e = 0; e < 8; ++e) {
                int row = rowBase + m * 16 + half * 8 + e;
                size_t idx = (size_t)row * N + n;
                float v = ac.f[e] + bv;
                if (ADDC) v += C[idx];
                C[idx] = v;
            }
        }
    }
}

// ---------------------------------------------------------------------------
// Causal depthwise conv (D_CONV=4) + bias + SiLU. xz is (M, 1536); conv input
// is its first 768 columns. Outputs u (f32) and u_bf (bf16).
// ---------------------------------------------------------------------------
__global__ void k_conv_silu(const float* __restrict__ xz, const float* __restrict__ cw,
                            const float* __restrict__ cb, float* __restrict__ u,
                            unsigned short* __restrict__ ubf) {
    int idx = blockIdx.x * blockDim.x + threadIdx.x;
    if (idx >= MM * DIN) return;
    int row = idx / DIN, d = idx - row * DIN;
    int b = row >> 9, l = row & (LL - 1);
    float acc = cb[d];
    #pragma unroll
    for (int t = 0; t < DCONV; ++t) {
        int lp = l + t - (DCONV - 1);
        if (lp >= 0)
            acc += xz[((size_t)(b * LL + lp)) * (2 * DIN) + d] * cw[d * DCONV + t];
    }
    float sv = acc * (1.0f / (1.0f + __expf(-acc)));   // SiLU
    u[idx] = sv;
    ubf[idx] = f2bf(sv);
}

// ---------------------------------------------------------------------------
// delta = softplus(delta_raw + dt_proj_b), in place.
// ---------------------------------------------------------------------------
__global__ void k_softplus(float* __restrict__ delta, const float* __restrict__ dtb) {
    int idx = blockIdx.x * blockDim.x + threadIdx.x;
    if (idx >= MM * DIN) return;
    float x = delta[idx] + dtb[idx % DIN];
    delta[idx] = (x > 20.0f) ? x : log1pf(__expf(x));
}

// ---------------------------------------------------------------------------
// Selective scan. grid=(DIN/256, B). Each thread owns one channel d of one
// batch: 16 states in registers; B/C (16 each) staged in LDS per timestep.
// Fuses +u*D, gate y*silu(res) and bf16 output for out_proj.
// ---------------------------------------------------------------------------
__global__ __launch_bounds__(256) void k_scan(
    const float* __restrict__ delta, const float* __restrict__ u,
    const float* __restrict__ xdbl, const float* __restrict__ xz,
    const float* __restrict__ A_log, const float* __restrict__ Dw,
    unsigned short* __restrict__ og) {
    __shared__ float sB[NST], sC[NST];
    int d = blockIdx.x * 256 + threadIdx.x;
    int b = blockIdx.y;
    float h[NST], An[NST];
    #pragma unroll
    for (int n = 0; n < NST; ++n) {
        h[n] = 0.0f;
        An[n] = -__expf(A_log[d * NST + n]);
    }
    float Dv = Dw[d];
    for (int l = 0; l < LL; ++l) {
        int row = b * LL + l;
        if (threadIdx.x < 32) {
            float v = xdbl[(size_t)row * XDBL_P + DTR + threadIdx.x];
            if (threadIdx.x < NST) sB[threadIdx.x] = v;
            else                   sC[threadIdx.x - NST] = v;
        }
        __syncthreads();
        float dv = delta[(size_t)row * DIN + d];
        float uv = u[(size_t)row * DIN + d];
        float du = dv * uv;
        float y = 0.0f;
        #pragma unroll
        for (int n = 0; n < NST; ++n) {
            float dA = __expf(dv * An[n]);
            h[n] = dA * h[n] + du * sB[n];
            y += h[n] * sC[n];
        }
        y += uv * Dv;
        float r = xz[(size_t)row * (2 * DIN) + DIN + d];
        float g = y * (r * (1.0f / (1.0f + __expf(-r))));
        og[(size_t)row * DIN + d] = f2bf(g);
        __syncthreads();
    }
}

// ---------------------------------------------------------------------------
extern "C" void kernel_launch(void* const* d_in, const int* in_sizes, int n_in,
                              void* d_out, int out_size, void* d_ws, size_t ws_size,
                              hipStream_t stream) {
    const float* input_ids = (const float*)d_in[0];
    const float* fc_W      = (const float*)d_in[1];
    const float* fc_b      = (const float*)d_in[2];
    const float* in_proj_W = (const float*)d_in[3];
    const float* conv_W    = (const float*)d_in[4];
    const float* conv_b    = (const float*)d_in[5];
    const float* x_proj_W  = (const float*)d_in[6];
    const float* dt_proj_W = (const float*)d_in[7];
    const float* dt_proj_b = (const float*)d_in[8];
    const float* A_log     = (const float*)d_in[9];
    const float* Dp        = (const float*)d_in[10];
    const float* out_proj_W= (const float*)d_in[11];
    const float* norm_W    = (const float*)d_in[12];
    const float* normf_W   = (const float*)d_in[13];
    const float* head_W    = (const float*)d_in[14];

    // Bump allocator over d_ws (256B aligned).
    char* base = (char*)d_ws;
    size_t off = 0;
    auto alloc = [&](size_t bytes) -> void* {
        void* p = base + off;
        off = (off + bytes + 255) & ~(size_t)255;
        return p;
    };
    unsigned short* in_bf      = (unsigned short*)alloc((size_t)MM * NVOCAB_P * 2);
    unsigned short* fcW_bf     = (unsigned short*)alloc((size_t)DMODEL * NVOCAB_P * 2);
    unsigned short* inprojW_bf = (unsigned short*)alloc((size_t)NLAYER * 2 * DIN * DMODEL * 2);
    unsigned short* xprojW_bf  = (unsigned short*)alloc((size_t)NLAYER * XDBL_P * DIN * 2);
    unsigned short* dtprojW_bf = (unsigned short*)alloc((size_t)NLAYER * DIN * DTR_P * 2);
    unsigned short* outprojW_bf= (unsigned short*)alloc((size_t)NLAYER * DMODEL * DIN * 2);
    unsigned short* headW_bf   = (unsigned short*)alloc((size_t)NFEAT * DMODEL * 2);
    float*          s_x        = (float*)alloc((size_t)MM * DMODEL * 4);
    unsigned short* s_h_bf     = (unsigned short*)alloc((size_t)MM * DMODEL * 2);
    float*          s_xz       = (float*)alloc((size_t)MM * 2 * DIN * 4);
    float*          s_u        = (float*)alloc((size_t)MM * DIN * 4);
    unsigned short* s_u_bf     = (unsigned short*)alloc((size_t)MM * DIN * 2);
    float*          s_xdbl     = (float*)alloc((size_t)MM * XDBL_P * 4);
    unsigned short* s_dt_bf    = (unsigned short*)alloc((size_t)MM * DTR_P * 2);
    float*          s_delta    = (float*)alloc((size_t)MM * DIN * 4);
    unsigned short* s_og_bf    = (unsigned short*)alloc((size_t)MM * DIN * 2);

    auto g1 = [](long n) { return (unsigned)((n + 255) / 256); };

    // --- weight / input conversion to bf16 (K padded to 64-multiples) ---
    k_cvt_pad<<<g1((long)MM * NVOCAB_P), 256, 0, stream>>>(input_ids, in_bf, MM, NVOCAB, NVOCAB_P, NVOCAB);
    k_cvt_pad<<<g1((long)DMODEL * NVOCAB_P), 256, 0, stream>>>(fc_W, fcW_bf, DMODEL, NVOCAB, NVOCAB_P, NVOCAB);
    k_cvt_pad<<<g1((long)NLAYER * 2 * DIN * DMODEL), 256, 0, stream>>>(in_proj_W, inprojW_bf, NLAYER * 2 * DIN, DMODEL, DMODEL, DMODEL);
    // x_proj: pad rows 56 -> 64 per layer with zeros (guard-free N=64 GEMM).
    k_cvt_pad<<<g1((long)NLAYER * XDBL_P * DIN), 256, 0, stream>>>(nullptr, xprojW_bf, NLAYER * XDBL_P, 0, DIN, DIN);
    for (int i = 0; i < NLAYER; ++i)
        k_cvt_pad<<<g1((long)XDBL * DIN), 256, 0, stream>>>(
            x_proj_W + (size_t)i * XDBL * DIN, xprojW_bf + (size_t)i * XDBL_P * DIN, XDBL, DIN, DIN, DIN);
    k_cvt_pad<<<g1((long)NLAYER * DIN * DTR_P), 256, 0, stream>>>(dt_proj_W, dtprojW_bf, NLAYER * DIN, DTR, DTR_P, DTR);
    k_cvt_pad<<<g1((long)NLAYER * DMODEL * DIN), 256, 0, stream>>>(out_proj_W, outprojW_bf, NLAYER * DMODEL, DIN, DIN, DIN);
    k_cvt_pad<<<g1((long)NFEAT * DMODEL), 256, 0, stream>>>(head_W, headW_bf, NFEAT, DMODEL, DMODEL, DMODEL);

    const dim3 blk(256);
    // --- fc: x = input @ fc_W^T + fc_b   (N=384, K=1600) ---
    k_gemm<2, 4, false><<<dim3(DMODEL / 128, MM / 128), blk, 0, stream>>>(in_bf, fcW_bf, s_x, fc_b, DMODEL, NVOCAB_P);

    for (int i = 0; i < NLAYER; ++i) {
        k_rmsnorm_bf<<<MM, 128, 0, stream>>>(s_x, norm_W + i * DMODEL, s_h_bf);
        // in_proj: xz = h @ W^T   (N=1536, K=384)
        k_gemm<2, 4, false><<<dim3(2 * DIN / 128, MM / 128), blk, 0, stream>>>(
            s_h_bf, inprojW_bf + (size_t)i * 2 * DIN * DMODEL, s_xz, nullptr, 2 * DIN, DMODEL);
        // depthwise causal conv + SiLU
        k_conv_silu<<<g1((long)MM * DIN), blk, 0, stream>>>(
            s_xz, conv_W + i * DIN * DCONV, conv_b + i * DIN, s_u, s_u_bf);
        // x_proj: xdbl = u @ W^T  (N=64 padded, K=768)
        k_gemm<2, 2, false><<<dim3(1, MM / 128), blk, 0, stream>>>(
            s_u_bf, xprojW_bf + (size_t)i * XDBL_P * DIN, s_xdbl, nullptr, XDBL_P, DIN);
        // dt slice -> bf16 padded (M,64)
        k_cvt_pad<<<g1((long)MM * DTR_P), 256, 0, stream>>>(s_xdbl, s_dt_bf, MM, DTR, DTR_P, XDBL_P);
        // dt_proj: delta_raw = dt @ W^T  (N=768, K=64)
        k_gemm<2, 4, false><<<dim3(DIN / 128, MM / 128), blk, 0, stream>>>(
            s_dt_bf, dtprojW_bf + (size_t)i * DIN * DTR_P, s_delta, nullptr, DIN, DTR_P);
        k_softplus<<<g1((long)MM * DIN), blk, 0, stream>>>(s_delta, dt_proj_b + i * DIN);
        // selective scan + gate
        k_scan<<<dim3(DIN / 256, BB), blk, 0, stream>>>(
            s_delta, s_u, s_xdbl, s_xz, A_log + (size_t)i * DIN * NST, Dp + i * DIN, s_og_bf);
        // out_proj with residual: x += og @ W^T  (N=384, K=768)
        k_gemm<2, 4, true><<<dim3(DMODEL / 128, MM / 128), blk, 0, stream>>>(
            s_og_bf, outprojW_bf + (size_t)i * DMODEL * DIN, s_x, nullptr, DMODEL, DIN);
    }

    // --- final norm + head ---
    k_rmsnorm_bf<<<MM, 128, 0, stream>>>(s_x, normf_W, s_h_bf);
    k_gemm<2, 4, false><<<dim3(NFEAT / 128, MM / 128), blk, 0, stream>>>(
        s_h_bf, headW_bf, (float*)d_out, nullptr, NFEAT, DMODEL);
    (void)in_sizes; (void)n_in; (void)out_size; (void)ws_size;
}
